// Paired_Windows_Attention_49452253446360
// MI455X (gfx1250) — compile-verified, matrix operand order checked by reading, not run on hardware
//
#include <hip/hip_runtime.h>
#include <hip/hip_bf16.h>

typedef __attribute__((ext_vector_type(16))) _Float16 v16h;
typedef __attribute__((ext_vector_type(8)))  float    v8f;

constexpr int  kB     = 4;
constexpr int  kC     = 64;
constexpr int  kH     = 48;
constexpr int  kHeads = 2;
constexpr int  kCph   = 8;     // channels per head per scale
constexpr int  kL     = 27;    // tokens per big window (3^3)
constexpr int  kSumN  = 4680;  // 16^3 + 8^3 + 4^3 + 2^3
constexpr int  kNrest = 584;   // windows in scales 1..3 (512+64+8)
constexpr long kPO    = (long)kB * kHeads * kSumN * kL * kCph; // pooled elems/tensor
constexpr int  kNW    = kB * kHeads * kSumN;                   // 37,440 windows
constexpr int  kVox   = kH * kH * kH;                          // 110,592

// ---------------------------------------------------------------------------
// Kernel 1a: scale 0 (sw=1) gather = pure relayout. Input-major mapping:
// lane == x  -> fully coalesced f32 reads (the dominant HBM stream).
// Scattered f16 writes land in the L2-resident pooled buffer.
// ---------------------------------------------------------------------------
__global__ void pool_s0_kernel(const float* __restrict__ q,
                               const float* __restrict__ k,
                               const float* __restrict__ v,
                               _Float16* __restrict__ pooled) {
  long idx = (long)blockIdx.x * blockDim.x + threadIdx.x;   // exact grid
  int x  = (int)(idx % kH);
  int y  = (int)((idx / kH) % kH);
  int z  = (int)((idx / (kH * kH)) % kH);
  int ch = (int)((idx / kVox) % 16);        // scale-0 channels are 0..15
  int b  = (int)((idx / ((long)kVox * 16)) % kB);
  int t  = (int)(idx / ((long)kVox * 16 * kB));

  const float* src = (t == 0 ? q : (t == 1 ? k : v));
  float val = src[(long)(b * kC + ch) * kVox + (z * kH + y) * kH + x];

  int head = ch >> 3, c = ch & 7;
  int Wz = z / 3, tz = z % 3;
  int Wy = y / 3, ty = y % 3;
  int Wx = x / 3, tx = x % 3;
  int n = (Wz * 16 + Wy) * 16 + Wx;
  int l = tz * 9 + ty * 3 + tx;
  long off = (((((long)t * kB + b) * kHeads + head) * kSumN + n) * kL + l) * kCph + c;
  pooled[off] = (_Float16)val;
}

// ---------------------------------------------------------------------------
// Kernel 1b: scales 1..3 gather + max-pool (output-major; few threads, each
// reduces sw^3 inputs).
// ---------------------------------------------------------------------------
__global__ void pool_rest_kernel(const float* __restrict__ q,
                                 const float* __restrict__ k,
                                 const float* __restrict__ v,
                                 _Float16* __restrict__ pooled) {
  long idx = (long)blockIdx.x * blockDim.x + threadIdx.x;   // exact grid
  int  c    = (int)(idx & 7);
  int  l    = (int)((idx >> 3) % kL);
  long rest = idx / (kL * kCph);
  int  nr   = (int)(rest % kNrest); rest /= kNrest;
  int  head = (int)(rest % kHeads); rest /= kHeads;
  int  b    = (int)(rest % kB);
  int  t    = (int)(rest / kB);
  int  n    = 4096 + nr;

  int s, wn;
  if      (nr < 512) { s = 1; wn = nr;       }
  else if (nr < 576) { s = 2; wn = nr - 512; }
  else               { s = 3; wn = nr - 576; }
  const int BWs[4] = {3, 6, 12, 24};
  const int SWs[4] = {1, 2, 4, 8};
  const int N1s[4] = {16, 8, 4, 2};
  int bw = BWs[s], sw = SWs[s], N1 = N1s[s];

  int Wx = wn % N1, Wy = (wn / N1) % N1, Wz = wn / (N1 * N1);
  int tx = l % 3,   ty = (l / 3) % 3,   tz = l / 9;
  int ch = (s * kHeads + head) * kCph + c;

  const float* src = (t == 0 ? q : (t == 1 ? k : v)) + (long)(b * kC + ch) * kVox;
  int z0 = Wz * bw + tz * sw;
  int y0 = Wy * bw + ty * sw;
  int x0 = Wx * bw + tx * sw;

  float mx = -3.4e38f;
  for (int pz = 0; pz < sw; ++pz)
    for (int py = 0; py < sw; ++py) {
      const float* row = src + ((long)(z0 + pz) * kH + (y0 + py)) * kH + x0;
      for (int px = 0; px < sw; ++px) mx = fmaxf(mx, row[px]);
    }
  long off = (((((long)t * kB + b) * kHeads + head) * kSumN + n) * kL + l) * kCph + c;
  pooled[off] = (_Float16)mx;
}

// ---------------------------------------------------------------------------
// Kernel 2: per-window attention, one wave32 per window.
//   QK^T: 2x2 tiles of v_wmma_f32_16x16x32_f16 (K=8 zero-padded to 32)
//   softmax: scores -> LDS (f32), one lane per row (no cross-lane shuffles),
//            post-softmax bias from LDS-staged rpb, W row -> LDS as f16
//   A*V: 2x1 tiles of v_wmma_f32_16x16x32_f16
// ---------------------------------------------------------------------------
__global__ void __launch_bounds__(128)
attn_kernel(const _Float16* __restrict__ qh, const _Float16* __restrict__ kh,
            const _Float16* __restrict__ vh, const float* __restrict__ rpb,
            float* __restrict__ attn) {
  __shared__ __align__(16) float    sbuf[4][32][36]; // 144B rows (16B aligned)
  __shared__ __align__(16) _Float16 wl[4][32][40];   // 80B rows (16B aligned)
  __shared__ float rpbs[250];                        // (125, HEADS) bias table

  // cooperative rpb stage (covered by the first __syncthreads below)
  for (int i = threadIdx.x; i < 250; i += 128) rpbs[i] = rpb[i];

  const int lane   = threadIdx.x & 31;
  const int wv     = threadIdx.x >> 5;
  const int wid    = blockIdx.x * 4 + wv;        // grid sized exactly: no guard
  const int head   = (wid / kSumN) % kHeads;
  const long base  = (long)wid * (kL * kCph);    // (b,head,n) contiguous-major
  const int half16 = lane >> 4;
  const int lr     = lane & 15;

  // ---- A fragments from Q (rows), B fragments from K (cols) ----
  v16h aq[2], bk[2];
#pragma unroll
  for (int tm = 0; tm < 2; ++tm) {
    v16h a;
#pragma unroll
    for (int j = 0; j < 16; ++j) a[j] = (_Float16)0;
    int m = tm * 16 + lr;
    if (half16 == 0 && m < kL) {
      const _Float16* p = qh + base + m * kCph;
#pragma unroll
      for (int j = 0; j < 8; ++j) a[j] = p[j];   // K=0..7 real, rest zero-pad
    }
    aq[tm] = a;
  }
#pragma unroll
  for (int tn = 0; tn < 2; ++tn) {
    v16h bfrag;
#pragma unroll
    for (int j = 0; j < 16; ++j) bfrag[j] = (_Float16)0;
    int n = tn * 16 + lr;
    if (half16 == 0 && n < kL) {
      const _Float16* p = kh + base + n * kCph;
#pragma unroll
      for (int j = 0; j < 8; ++j) bfrag[j] = p[j];
    }
    bk[tn] = bfrag;
  }

  // ---- scores: 4 WMMAs (EXEC all-ones) ----
  v8f sc[2][2];
#pragma unroll
  for (int tm = 0; tm < 2; ++tm)
#pragma unroll
    for (int tn = 0; tn < 2; ++tn) {
      v8f cz = {};
      sc[tm][tn] = __builtin_amdgcn_wmma_f32_16x16x32_f16(
          false, aq[tm], false, bk[tn], (short)0, cz, false, false);
    }

  // ---- spill raw scores to LDS: lane holds rows 16*tm+8*half16+r, cols lr,16+lr
#pragma unroll
  for (int tm = 0; tm < 2; ++tm)
#pragma unroll
    for (int r = 0; r < 8; ++r) {
      int m = tm * 16 + half16 * 8 + r;
      sbuf[wv][m][lr]      = sc[tm][0][r];
      sbuf[wv][m][16 + lr] = sc[tm][1][r];
    }
  __syncthreads();

  // ---- per-row softmax + post-softmax bias: one lane per row, no shuffles ---
  const float rs = 0.35355339059327373f;  // 1/sqrt(8)
  if (lane < kL) {
    const int m = lane;
    const float* row = &sbuf[wv][m][0];
    float e[kL];
    float mx = -3.4e38f;
#pragma unroll
    for (int n = 0; n < kL; ++n) { e[n] = row[n] * rs; mx = fmaxf(mx, e[n]); }
    float su = 0.f;
#pragma unroll
    for (int n = 0; n < kL; ++n) { e[n] = __expf(e[n] - mx); su += e[n]; }
    float inv = 1.f / su;
    int mz = m / 9, my = (m / 3) % 3, mxi = m % 3;
    int bidx0 = ((mz + 2) * 25 + (my + 2) * 5 + (mxi + 2)) * kHeads + head;
    int n = 0;
#pragma unroll
    for (int nz = 0; nz < 3; ++nz)
#pragma unroll
      for (int ny = 0; ny < 3; ++ny)
#pragma unroll
        for (int nx = 0; nx < 3; ++nx, ++n) {
          float w = e[n] * inv + rpbs[bidx0 - (nz * 25 + ny * 5 + nx) * kHeads];
          wl[wv][m][n] = (_Float16)w;
        }
#pragma unroll
    for (int p = kL; p < 32; ++p) wl[wv][m][p] = (_Float16)0;  // pad cols
  } else {
#pragma unroll
    for (int p = 0; p < 32; ++p) wl[wv][lane][p] = (_Float16)0; // pad rows 27..31
  }
  __syncthreads();

  // ---- B fragment from V (cols = 8 channels, K = 27 tokens padded to 32) ----
  v16h bv;
#pragma unroll
  for (int j = 0; j < 16; ++j) bv[j] = (_Float16)0;
  if (lr < kCph) {
#pragma unroll
    for (int j = 0; j < 16; ++j) {
      int K = (j < 8) ? (half16 * 8 + j) : (16 + half16 * 8 + (j - 8));
      if (K < kL) bv[j] = vh[base + K * kCph + lr];
    }
  }

  // ---- A fragments from W (LDS), then out = W*V: 2 WMMAs ----
  v8f o[2];
#pragma unroll
  for (int tm = 0; tm < 2; ++tm) {
    const _Float16* wrow = &wl[wv][tm * 16 + lr][0];
    v16h aw;
#pragma unroll
    for (int jj = 0; jj < 8; ++jj) {             // paired (even K) 4B loads
      int K2 = half16 * 8 + (jj < 4 ? 2 * jj : 16 + 2 * (jj - 4));
      aw[2 * jj]     = wrow[K2];
      aw[2 * jj + 1] = wrow[K2 + 1];
    }
    v8f cz = {};
    o[tm] = __builtin_amdgcn_wmma_f32_16x16x32_f16(
        false, aw, false, bv, (short)0, cz, false, false);
  }

  // ---- store 27x8 f32 per window ----
#pragma unroll
  for (int tm = 0; tm < 2; ++tm)
#pragma unroll
    for (int r = 0; r < 8; ++r) {
      int m = tm * 16 + half16 * 8 + r;
      if (m < kL && lr < kCph) attn[base + m * kCph + lr] = o[tm][r];
    }
}

// ---------------------------------------------------------------------------
// Kernel 3: trilinear upsample (align_corners) back to (B, C, 48^3)
// ---------------------------------------------------------------------------
__global__ void scatter_kernel(const float* __restrict__ attn,
                               float* __restrict__ out) {
  long idx = (long)blockIdx.x * blockDim.x + threadIdx.x;   // exact grid
  int x  = (int)(idx % kH);
  int y  = (int)((idx / kH) % kH);
  int z  = (int)((idx / (kH * kH)) % kH);
  int ch = (int)((idx / (long)kVox) % kC);
  int b  = (int)(idx / ((long)kC * kVox));

  int s = ch >> 4, head = (ch >> 3) & 1, c = ch & 7;
  const int BWs[4] = {3, 6, 12, 24};
  const int N1s[4] = {16, 8, 4, 2};
  const int OFF[4] = {0, 4096, 4608, 4672};
  int bw = BWs[s], N1 = N1s[s];

  int Wz = z / bw, Wy = y / bw, Wx = x / bw;
  int pz = z % bw, py = y % bw, px = x % bw;
  int n = OFF[s] + (Wz * N1 + Wy) * N1 + Wx;
  long base = ((long)(b * kHeads + head) * kSumN + n) * (kL * kCph) + c;

  float sc2 = 2.0f / (float)(bw - 1);           // (n_in-1)/(n_out-1), n_in=3
  float fz = pz * sc2; int iz = (int)fz; if (iz > 1) iz = 1; float wz = fz - iz;
  float fy = py * sc2; int iy = (int)fy; if (iy > 1) iy = 1; float wy = fy - iy;
  float fx = px * sc2; int ix = (int)fx; if (ix > 1) ix = 1; float wx = fx - ix;

  float res = 0.f;
#pragma unroll
  for (int dz = 0; dz < 2; ++dz)
#pragma unroll
    for (int dy = 0; dy < 2; ++dy)
#pragma unroll
      for (int dx = 0; dx < 2; ++dx) {
        float w = (dz ? wz : 1.f - wz) * (dy ? wy : 1.f - wy) *
                  (dx ? wx : 1.f - wx);
        int l = ((iz + dz) * 3 + (iy + dy)) * 3 + (ix + dx);
        res += w * attn[base + (long)l * kCph];
      }
  out[idx] = res;
}

// ---------------------------------------------------------------------------
extern "C" void kernel_launch(void* const* d_in, const int* in_sizes, int n_in,
                              void* d_out, int out_size, void* d_ws, size_t ws_size,
                              hipStream_t stream) {
  const float* q   = (const float*)d_in[0];
  const float* k   = (const float*)d_in[1];
  const float* v   = (const float*)d_in[2];
  const float* rpb = (const float*)d_in[3];
  float* out = (float*)d_out;

  _Float16* pooled = (_Float16*)d_ws;                       // 3 * kPO f16
  float* attnb = (float*)((char*)d_ws + 3L * kPO * sizeof(_Float16)); // kPO f32

  long totS0 = 3L * kB * 16 * kVox;          // 21,233,664 -> /256 exact
  pool_s0_kernel<<<(int)(totS0 / 256), 256, 0, stream>>>(q, k, v, pooled);

  long totR = 3L * kB * kHeads * kNrest * kL * kCph; // 3,027,456 -> /256 exact
  pool_rest_kernel<<<(int)(totR / 256), 256, 0, stream>>>(q, k, v, pooled);

  attn_kernel<<<kNW / 4, 128, 0, stream>>>(pooled, pooled + kPO,
                                           pooled + 2 * kPO, rpb, attnb);

  long totOut = (long)kB * kC * kVox;        // 28,311,552 -> /256 exact
  scatter_kernel<<<(int)(totOut / 256), 256, 0, stream>>>(attnb, out);
}
